// FCGF_point_att3_fc_89575837925665
// MI455X (gfx1250) — matrix operand order repaired and checked by reading, not compiled, not run on hardware
//
#include <hip/hip_runtime.h>
#include <hip/hip_bf16.h>
#include <math.h>

typedef __attribute__((ext_vector_type(16))) _Float16 v16h;
typedef __attribute__((ext_vector_type(8)))  float    v8f;

namespace {
constexpr int   N_PER = 62500;          // points per segment
constexpr int   SEGS  = 32;             // segments
constexpr int   BPS   = 32;             // blocks per segment
constexpr int   NW    = 8;              // waves per block (256 threads)
constexpr int   NT    = (N_PER + 15) / 16;  // 16-point tiles per segment
constexpr int   REC   = 34;             // floats per block record: m, esum, xsum[32]
constexpr int   STF   = 18;             // per-thread state floats: m, esum, acc[16]
constexpr float kEps  = 1e-5f;
}

// ---------------------------------------------------------------------------
// Kernel 1: fused score (WMMA) + per-block online-softmax pooling partials.
//
// Transposed mapping: D = w1 @ x^T via v_wmma_f32_16x16x32_f16.
//   A = w1   (M=16 channels x K=32 features, f16)  -> loaded once
//   B = x^T  (K=32 features x N=16 points,  f16)   -> lane ell holds point
//            p = ell&15, features kb2..kb2+15 (kb2 = 16*(ell>=16)): one
//            contiguous 64B chunk per lane, 2KB contiguous per tile.
//   D: lane ell holds point p = ell&15, channels r + 8*(ell>=16).
// Each lane owns one point per tile: BN1+ReLU+w2 dot = 8 in-lane FMAs, then a
// single shfl_xor(.,16) adds the channel halves -> score a_p. Each lane runs a
// lane-local online softmax (m, sum e, sum x[f]*e over its 16 features); the
// 256 lane states are merged once per block in LDS in fixed order
// (deterministic), producing one 34-float record in ws.
// ---------------------------------------------------------------------------
__global__ __launch_bounds__(256) void fcgf_score_pool(
    const float* __restrict__ x,
    const float* __restrict__ w1, const float* __restrict__ b1,
    const float* __restrict__ g1, const float* __restrict__ be1,
    const float* __restrict__ m1, const float* __restrict__ v1,
    const float* __restrict__ w2, const float* __restrict__ b2,
    const float* __restrict__ g2, const float* __restrict__ be2,
    const float* __restrict__ m2, const float* __restrict__ v2,
    float* __restrict__ ws)
{
  const int lane = threadIdx.x & 31;
  const int wid  = threadIdx.x >> 5;
  const int seg  = blockIdx.y;
  const int blk  = blockIdx.x;
  const int pcol = lane & 15;          // point column owned by this lane
  const bool hi  = lane >= 16;

  // BN2 fold (scalar) and per-lane constants for the 8 channels in D rows.
  const float s2 = g2[0] * rsqrtf(v2[0] + kEps);
  const float ab = (b2[0] - m2[0]) * s2 + be2[0];
  const int chb = hi ? 8 : 0;
  float s1v[8], hb1v[8], w2v[8];
  #pragma unroll
  for (int r = 0; r < 8; ++r) {
    const int ch = chb + r;
    s1v[r]  = g1[ch] * rsqrtf(v1[ch] + kEps);
    hb1v[r] = (b1[ch] - m1[ch]) * s1v[r] + be1[ch];
    w2v[r]  = w2[ch] * s2;
  }

  // A matrix = w1 (16x32 f16), loaded once. Lane: row m=pcol (channel),
  // K = kbase..kbase+15, kbase = 16*(hi).
  const int kbase = hi ? 16 : 0;
  v16h am;
  #pragma unroll
  for (int j = 0; j < 8; ++j) {
    am[j]     = (_Float16)w1[pcol * 32 + kbase + j];
    am[8 + j] = (_Float16)w1[pcol * 32 + kbase + 8 + j];
  }

  const int kb2 = hi ? 16 : 0;         // B-matrix K base == feature base

  // Lane-local online softmax state over this lane's point stream.
  float m_l  = -INFINITY;
  float esum = 0.f;
  float acc[16];
  #pragma unroll
  for (int j = 0; j < 16; ++j) acc[j] = 0.f;

  const int wstride = BPS * NW;
  const float* xseg = x + (size_t)seg * N_PER * 32;
  for (int t = blk * NW + wid; t < NT; t += wstride) {
    const int p  = t * 16 + pcol;                    // point index in segment
    const int rp = (p < N_PER) ? p : (N_PER - 1);    // clamp tail (masked below)
    const float* xr = xseg + (size_t)rp * 32 + kb2;  // 64B contiguous per lane

    // Prefetch next tile's chunk (global_prefetch_b8).
    {
      const int tn  = t + wstride;
      const int pn  = tn * 16 + pcol;
      const int rpn = (pn < N_PER) ? pn : (N_PER - 1);
      __builtin_prefetch(xseg + (size_t)rpn * 32 + kb2, 0, 1);
    }

    const float4* q = (const float4*)xr;
    const float4 q0 = q[0], q1 = q[1], q2 = q[2], q3 = q[3];
    const float xf[16] = {q0.x,q0.y,q0.z,q0.w, q1.x,q1.y,q1.z,q1.w,
                          q2.x,q2.y,q2.z,q2.w, q3.x,q3.y,q3.z,q3.w};
    v16h bm;
    #pragma unroll
    for (int e = 0; e < 16; ++e) bm[e] = (_Float16)xf[e];

    const v8f cz = (v8f)(0.0f);
    const v8f d = __builtin_amdgcn_wmma_f32_16x16x32_f16(
        false, am, false, bm, (short)0, cz, false, false);

    // BN1 + ReLU + w2-weighted partial dot over this lane's 8 channels.
    float pa = 0.f;
    #pragma unroll
    for (int r = 0; r < 8; ++r)
      pa = fmaf(fmaxf(d[r] * s1v[r] + hb1v[r], 0.f), w2v[r], pa);
    // Add the other channel half: lanes ell and ell^16 hold the same point.
    const float a = pa + __shfl_xor(pa, 16, 32) + ab;

    // Lane-local online softmax update (branchless).
    const float newm = fmaxf(m_l, a);
    const float sc   = __expf(m_l - newm);           // exp(-inf)=0 first time
    const float e    = (p < N_PER) ? __expf(a - newm) : 0.f;
    m_l  = newm;
    esum = esum * sc + e;
    #pragma unroll
    for (int j = 0; j < 16; ++j) acc[j] = acc[j] * sc + xf[j] * e;
  }

  // Deterministic block merge of 256 lane-local states via LDS.
  __shared__ float s_st[NW * 32][STF];
  const int tid = threadIdx.x;
  s_st[tid][0] = m_l;
  s_st[tid][1] = esum;
  #pragma unroll
  for (int j = 0; j < 16; ++j) s_st[tid][2 + j] = acc[j];
  __syncthreads();

  if (tid < 32) {
    float m_blk = -INFINITY;
    for (int th = 0; th < NW * 32; ++th) m_blk = fmaxf(m_blk, s_st[th][0]);

    // Feature f: owned by lanes (f<16 ? 0..15 : 16..31) of every wave;
    // in-lane slot j = f & 15.
    const int f    = tid;
    const int lsel = (f < 16) ? 0 : 16;
    const int j    = f & 15;
    float xs = 0.f;
    for (int w = 0; w < NW; ++w)
      for (int l2 = 0; l2 < 16; ++l2) {
        const int th = w * 32 + lsel + l2;
        xs += s_st[th][2 + j] * __expf(s_st[th][0] - m_blk);
      }
    float* rec = ws + (size_t)(seg * BPS + blk) * REC;
    rec[2 + f] = xs;
    if (f == 0) {
      // esum duplicated across lane halves -> sum lanes 0..15 only (exact).
      float es = 0.f;
      for (int w = 0; w < NW; ++w)
        for (int l2 = 0; l2 < 16; ++l2) {
          const int th = w * 32 + l2;
          es += s_st[th][1] * __expf(s_st[th][0] - m_blk);
        }
      rec[0] = m_blk;
      rec[1] = es;
    }
  }
}

// ---------------------------------------------------------------------------
// Kernel 2 (tiny, one block): merge per-block softmax records -> pooled[32,32],
// FC head (32->64->256) with BN folds, L2 row normalize. ~0.6 MFLOP total.
// ---------------------------------------------------------------------------
__global__ __launch_bounds__(256) void fcgf_head(
    const float* __restrict__ ws, const long long* __restrict__ len,
    const float* __restrict__ fw1, const float* __restrict__ fb1,
    const float* __restrict__ fg1, const float* __restrict__ fbe1,
    const float* __restrict__ fm1, const float* __restrict__ fv1,
    const float* __restrict__ fw2, const float* __restrict__ fb2,
    const float* __restrict__ fg2, const float* __restrict__ fbe2,
    const float* __restrict__ fm2, const float* __restrict__ fv2,
    float* __restrict__ out)
{
  __shared__ float pooled[SEGS * 32];
  __shared__ float h1[SEGS * 64];
  __shared__ float rr[SEGS * 256];
  __shared__ float inv[SEGS];
  const int lane = threadIdx.x & 31;
  const int wid  = threadIdx.x >> 5;

  // Merge records (one wave per segment, 4 segments per wave).
  for (int s = wid; s < SEGS; s += NW) {
    const float* base = ws + (size_t)s * BPS * REC;
    float pm = -INFINITY;
    for (int b = lane; b < BPS; b += 32) pm = fmaxf(pm, base[b * REC]);
    #pragma unroll
    for (int mk = 1; mk <= 16; mk <<= 1) pm = fmaxf(pm, __shfl_xor(pm, mk, 32));
    float pe = 0.f;
    for (int b = lane; b < BPS; b += 32)
      pe += base[b * REC + 1] * __expf(base[b * REC] - pm);
    #pragma unroll
    for (int mk = 1; mk <= 16; mk <<= 1) pe += __shfl_xor(pe, mk, 32);
    // Feature sums: lane == feature, fixed-order over blocks.
    float xs = 0.f;
    for (int b = 0; b < BPS; ++b)
      xs += base[b * REC + 2 + lane] * __expf(base[b * REC] - pm);
    pooled[s * 32 + lane] = xs / pe / (float)len[s];
  }
  __syncthreads();

  // FC1 + BN + ReLU -> h1[32,64]
  for (int idx = threadIdx.x; idx < SEGS * 64; idx += 256) {
    const int s = idx >> 6, o = idx & 63;
    float d = 0.f;
    #pragma unroll
    for (int k = 0; k < 32; ++k) d = fmaf(pooled[s * 32 + k], fw1[o * 32 + k], d);
    const float sc = fg1[o] * rsqrtf(fv1[o] + kEps);
    h1[idx] = fmaxf((d + fb1[o] - fm1[o]) * sc + fbe1[o], 0.f);
  }
  __syncthreads();

  // FC2 + BN -> rr[32,256]
  for (int idx = threadIdx.x; idx < SEGS * 256; idx += 256) {
    const int s = idx >> 8, o = idx & 255;
    float d = 0.f;
    #pragma unroll
    for (int k = 0; k < 64; ++k) d = fmaf(h1[s * 64 + k], fw2[o * 64 + k], d);
    const float sc = fg2[o] * rsqrtf(fv2[o] + kEps);
    rr[idx] = (d + fb2[o] - fm2[o]) * sc + fbe2[o];
  }
  __syncthreads();

  // L2 normalize rows
  if (threadIdx.x < SEGS) {
    const int s = threadIdx.x;
    float ss = 0.f;
    for (int k = 0; k < 256; ++k) { const float v = rr[s * 256 + k]; ss = fmaf(v, v, ss); }
    inv[s] = 1.f / fmaxf(sqrtf(ss), 1e-12f);
  }
  __syncthreads();
  for (int idx = threadIdx.x; idx < SEGS * 256; idx += 256)
    out[idx] = rr[idx] * inv[idx >> 8];
}

// ---------------------------------------------------------------------------
extern "C" void kernel_launch(void* const* d_in, const int* in_sizes, int n_in,
                              void* d_out, int out_size, void* d_ws, size_t ws_size,
                              hipStream_t stream) {
  const float*     x    = (const float*)d_in[0];
  const long long* len  = (const long long*)d_in[1];
  // d_in[2] segment_ids: unused (segments are contiguous, equal length)
  const float* w1  = (const float*)d_in[3];
  const float* b1  = (const float*)d_in[4];
  const float* g1  = (const float*)d_in[5];
  const float* be1 = (const float*)d_in[6];
  const float* m1  = (const float*)d_in[7];
  const float* v1  = (const float*)d_in[8];
  const float* w2  = (const float*)d_in[9];
  const float* b2  = (const float*)d_in[10];
  const float* g2  = (const float*)d_in[11];
  const float* be2 = (const float*)d_in[12];
  const float* m2  = (const float*)d_in[13];
  const float* v2  = (const float*)d_in[14];
  const float* fw1 = (const float*)d_in[15];
  const float* fb1 = (const float*)d_in[16];
  const float* fg1 = (const float*)d_in[17];
  const float* fbe1= (const float*)d_in[18];
  const float* fm1 = (const float*)d_in[19];
  const float* fv1 = (const float*)d_in[20];
  const float* fw2 = (const float*)d_in[21];
  const float* fb2 = (const float*)d_in[22];
  const float* fg2 = (const float*)d_in[23];
  const float* fbe2= (const float*)d_in[24];
  const float* fm2 = (const float*)d_in[25];
  const float* fv2 = (const float*)d_in[26];

  float* ws  = (float*)d_ws;           // needs 32*32*34*4 = 139,264 bytes
  float* out = (float*)d_out;

  dim3 grid(BPS, SEGS);
  fcgf_score_pool<<<grid, 256, 0, stream>>>(
      x, w1, b1, g1, be1, m1, v1, w2, b2, g2, be2, m2, v2, ws);
  fcgf_head<<<1, 256, 0, stream>>>(
      ws, len, fw1, fb1, fg1, fbe1, fm1, fv1,
      fw2, fb2, fg2, fbe2, fm2, fv2, out);
}
